// IzhikevichTwoComp_21294447854018
// MI455X (gfx1250) — matrix operand order repaired and checked by reading, not compiled, not run on hardware
//
#include <hip/hip_runtime.h>
#include <hip/hip_bf16.h>

// ---------------------------------------------------------------------------
// IzhikevichTwoComp rollout for MI455X (gfx1250, wave32).
//   FF = x_seq @ W_FF        [1024 x 4096] = [1024x4096][4096x4096]
//   FB = fb_seq @ W_FB       [1024 x 4096] = [1024x1024][1024x4096]
//   scan: elementwise Izhikevich recurrence over T using FF/FB/att,
//         double-buffered through LDS with global_load_async_to_lds_b128.
// GEMMs use v_wmma_f32_16x16x32_bf16 with a bf16 hi/lo split (3 products)
// for near-fp32 accuracy at 8x the f32-WMMA FLOP rate, with software-
// pipelined global->LDS staging (loads for chunk k+1 overlap chunk k wmma).
// ---------------------------------------------------------------------------

typedef __attribute__((ext_vector_type(16))) __bf16 v16bf;
typedef __attribute__((ext_vector_type(8)))  __bf16 v8bf;
typedef __attribute__((ext_vector_type(4)))  __bf16 v4bf;
typedef __attribute__((ext_vector_type(8)))  float  v8f;

#define TILE_M 128
#define TILE_N 128
#define TILE_K 32
#define LDK    40   // padded K stride (bf16 elems): 80 B rows -> 16B-aligned frags

static __device__ __forceinline__ v16bf combine8(v8bf lo, v8bf hi) {
    return __builtin_shufflevector(lo, hi, 0, 1, 2, 3, 4, 5, 6, 7,
                                           8, 9, 10, 11, 12, 13, 14, 15);
}

// Async 16-byte global -> LDS copy (CDNA5, tracked by ASYNCcnt).
// LDS byte address = low 32 bits of the generic shared-memory pointer
// (ISA 10.2: LDS aperture address = addr[31:0]).
static __device__ __forceinline__ void async_copy16(const float* g, float* l) {
    const unsigned lds = (unsigned)(uintptr_t)l;
    const unsigned long long ga = (unsigned long long)(uintptr_t)g;
    asm volatile("global_load_async_to_lds_b128 %0, %1, off"
                 :: "v"(lds), "v"(ga) : "memory");
}
static __device__ __forceinline__ void wait_async0() {
    asm volatile("s_wait_asynccnt 0x0" ::: "memory");
}

// ---------------------------------------------------------------------------
// C[M x N] = A[M x K] @ B[K x N]
// ---------------------------------------------------------------------------
__global__ __launch_bounds__(256)
void gemm_bf16x3(const float* __restrict__ A, const float* __restrict__ B,
                 float* __restrict__ C, int K, int N) {
    __shared__ __bf16 Ah[TILE_M * LDK];
    __shared__ __bf16 Al[TILE_M * LDK];
    __shared__ __bf16 Bh[TILE_N * LDK];
    __shared__ __bf16 Bl[TILE_N * LDK];

    const int tid  = threadIdx.x;
    const int lane = tid & 31;
    const int wid  = tid >> 5;
    const int hl   = lane >> 4;      // which 16-lane half
    const int nl   = lane & 15;
    const int wy   = wid & 3;        // wave row: 4 waves cover 128 M (32 each)
    const int wx   = wid >> 2;       // wave col: 2 waves cover 128 N (64 each)
    const int bm0  = blockIdx.y * TILE_M;
    const int bn0  = blockIdx.x * TILE_N;

    const int arow = tid >> 3;       // 0..31 (A row per pass)
    const int akc  = (tid & 7) * 4;  // A k-offset
    const int bk   = tid >> 5;       // 0..7 (B k per pass)
    const int bnc  = (tid & 31) * 4; // B n-offset

    v8f acc[2][4];
#pragma unroll
    for (int i = 0; i < 2; ++i)
#pragma unroll
        for (int j = 0; j < 4; ++j)
#pragma unroll
            for (int r = 0; r < 8; ++r) acc[i][j][r] = 0.0f;

    // ---- prologue: stage chunk 0 into registers ---------------------------
    float4 a4[4], b4[4];
#pragma unroll
    for (int p = 0; p < 4; ++p)
        a4[p] = *reinterpret_cast<const float4*>(
            &A[(size_t)(bm0 + p * 32 + arow) * K + akc]);
#pragma unroll
    for (int p = 0; p < 4; ++p)
        b4[p] = *reinterpret_cast<const float4*>(
            &B[(size_t)(p * 8 + bk) * N + bn0 + bnc]);

    for (int kb = 0; kb < K; kb += TILE_K) {
        __syncthreads();  // previous chunk's LDS consumers done

        // ---- commit staged registers to LDS as bf16 hi/lo -----------------
#pragma unroll
        for (int p = 0; p < 4; ++p) {
            const int row = p * 32 + arow;
            float va[4] = {a4[p].x, a4[p].y, a4[p].z, a4[p].w};
            v4bf h, l;
#pragma unroll
            for (int j = 0; j < 4; ++j) {
                __bf16 hh = (__bf16)va[j];
                h[j] = hh;
                l[j] = (__bf16)(va[j] - (float)hh);
            }
            *reinterpret_cast<v4bf*>(&Ah[row * LDK + akc]) = h;
            *reinterpret_cast<v4bf*>(&Al[row * LDK + akc]) = l;
        }
#pragma unroll
        for (int p = 0; p < 4; ++p) {
            const int k = p * 8 + bk;
            float vb[4] = {b4[p].x, b4[p].y, b4[p].z, b4[p].w};
#pragma unroll
            for (int j = 0; j < 4; ++j) {
                __bf16 hh = (__bf16)vb[j];
                Bh[(bnc + j) * LDK + k] = hh;
                Bl[(bnc + j) * LDK + k] = (__bf16)(vb[j] - (float)hh);
            }
        }

        __syncthreads();

        // ---- issue next chunk's global loads (overlap with wmma below) ----
        const int kn = kb + TILE_K;
        if (kn < K) {
#pragma unroll
            for (int p = 0; p < 4; ++p)
                a4[p] = *reinterpret_cast<const float4*>(
                    &A[(size_t)(bm0 + p * 32 + arow) * K + kn + akc]);
#pragma unroll
            for (int p = 0; p < 4; ++p)
                b4[p] = *reinterpret_cast<const float4*>(
                    &B[(size_t)(kn + p * 8 + bk) * N + bn0 + bnc]);
        }

        // ---- fragment loads (layouts per CDNA5 ISA 7.12.2) ----------------
        // A 16x32 bf16: lane half hl, row m=nl; e<8 -> K=hl*8+e ; e>=8 -> 16+hl*8+(e-8)
        v16bf fAh[2], fAl[2], fBh[4], fBl[4];
#pragma unroll
        for (int i = 0; i < 2; ++i) {
            const int row = wy * 32 + i * 16 + nl;
            fAh[i] = combine8(
                *reinterpret_cast<const v8bf*>(&Ah[row * LDK + hl * 8]),
                *reinterpret_cast<const v8bf*>(&Ah[row * LDK + 16 + hl * 8]));
            fAl[i] = combine8(
                *reinterpret_cast<const v8bf*>(&Al[row * LDK + hl * 8]),
                *reinterpret_cast<const v8bf*>(&Al[row * LDK + 16 + hl * 8]));
        }
        // B 32x16 bf16: lane half hl, col n=nl; e<8 -> K=hl*8+e ; e>=8 -> 16+hl*8+(e-8)
#pragma unroll
        for (int j = 0; j < 4; ++j) {
            const int rn = wx * 64 + j * 16 + nl;
            fBh[j] = combine8(
                *reinterpret_cast<const v8bf*>(&Bh[rn * LDK + hl * 8]),
                *reinterpret_cast<const v8bf*>(&Bh[rn * LDK + 16 + hl * 8]));
            fBl[j] = combine8(
                *reinterpret_cast<const v8bf*>(&Bl[rn * LDK + hl * 8]),
                *reinterpret_cast<const v8bf*>(&Bl[rn * LDK + 16 + hl * 8]));
        }

        // ---- 3-product bf16 WMMA: A*B ~= Al*Bh + Ah*Bl + Ah*Bh ------------
#pragma unroll
        for (int i = 0; i < 2; ++i)
#pragma unroll
            for (int j = 0; j < 4; ++j) {
                acc[i][j] = __builtin_amdgcn_wmma_f32_16x16x32_bf16(
                    false, fAl[i], false, fBh[j], (short)0, acc[i][j], false, false);
                acc[i][j] = __builtin_amdgcn_wmma_f32_16x16x32_bf16(
                    false, fAh[i], false, fBl[j], (short)0, acc[i][j], false, false);
                acc[i][j] = __builtin_amdgcn_wmma_f32_16x16x32_bf16(
                    false, fAh[i], false, fBh[j], (short)0, acc[i][j], false, false);
            }
    }

    // ---- epilogue: C/D layout -> VGPR r holds M = hl*8 + r, N = nl ---------
#pragma unroll
    for (int i = 0; i < 2; ++i)
#pragma unroll
        for (int j = 0; j < 4; ++j) {
            const int col = bn0 + wx * 64 + j * 16 + nl;
#pragma unroll
            for (int r = 0; r < 8; ++r) {
                const int row = bm0 + wy * 32 + i * 16 + hl * 8 + r;
                C[(size_t)row * N + col] = acc[i][j][r];
            }
        }
}

// ---------------------------------------------------------------------------
// Sequential Izhikevich scan: one thread per neuron, T steps.
// FF/FB/att are streamed through LDS with async b128 copies, double-buffered
// in chunks of CH timesteps so HBM/L2 latency hides under compute.
// ---------------------------------------------------------------------------
#define CH 8

__global__ __launch_bounds__(256)
void izh_scan(const float* __restrict__ att, const float* __restrict__ bias,
              const float* __restrict__ FF, const float* __restrict__ FB,
              float* __restrict__ out, int T, int N) {
    __shared__ float sFF[2][CH * 256];
    __shared__ float sFB[2][CH * 256];
    __shared__ float sAT[2][CH * 256];

    const int tid = threadIdx.x;
    const int n0  = blockIdx.x * 256;
    const int n   = n0 + tid;

    float v  = -65.0f;
    float u  = 0.2f * -65.0f;
    float va = 0.0f;
    float m  = 0.0f;
    const float b = bias[n];

    // issue async copies for one chunk (CH x 256 floats per stream)
    auto issue = [&](int t0, int buf) {
#pragma unroll
        for (int q = 0; q < 2; ++q) {
            const int flat = q * 256 + tid;   // 512 sixteen-byte units / stream
            const int row  = flat >> 6;       // 0..CH-1
            const int col  = (flat & 63) * 4; // float col 0..252
            const size_t go = (size_t)(t0 + row) * N + n0 + col;
            const int lo = row * 256 + col;
            async_copy16(&FF[go],  &sFF[buf][lo]);
            async_copy16(&FB[go],  &sFB[buf][lo]);
            async_copy16(&att[go], &sAT[buf][lo]);
        }
    };

    issue(0, 0);
    wait_async0();
    __syncthreads();

    const int nchunks = T / CH;
    for (int c = 0; c < nchunks; ++c) {
        const int buf = c & 1;
        if (c + 1 < nchunks) issue((c + 1) * CH, buf ^ 1);

#pragma unroll
        for (int r = 0; r < CH; ++r) {
            const float ff = sFF[buf][r * 256 + tid];
            const float fb = sFB[buf][r * 256 + tid];
            const float at = sAT[buf][r * 256 + tid];

            m = 0.9f * m + 0.1f * at;
            // apical: 0.5*(sigmoid(va)-0.5)*20 = 10*sigmoid(va) - 5
            const float sig = 1.0f / (1.0f + __expf(-va));
            const float I = ff * 8.0f + b + (10.0f * sig - 5.0f) + m;
#pragma unroll
            for (int h = 0; h < 2; ++h) {
                const float dv = 0.04f * v * v + 5.0f * v + 140.0f - u + I;
                const float du = 0.02f * (0.2f * v - u);
                v += 0.5f * dv;
                u += 0.5f * du;
            }
            if (v >= 30.0f) { v = -65.0f; u += 8.0f; }
            v = fminf(fmaxf(v, -100.0f), 35.0f);
            u = fminf(fmaxf(u, -100.0f), 100.0f);
            out[(size_t)(c * CH + r) * N + n] =
                fminf(fmaxf((v + 65.0f) * (1.0f / 95.0f), 0.0f), 1.0f);
            va = fminf(fmaxf(0.65f * va + 0.35f * fb, -5.0f), 5.0f);
        }

        wait_async0();   // next chunk's LDS writes complete (this wave)
        __syncthreads(); // publish across all waves before reuse
    }
}

extern "C" void kernel_launch(void* const* d_in, const int* in_sizes, int n_in,
                              void* d_out, int out_size, void* d_ws, size_t ws_size,
                              hipStream_t stream) {
    const int T = 1024, N_IN = 4096, N_OUT = 4096, N_FB = 1024;

    const float* x    = (const float*)d_in[0];  // [T, N_IN]
    const float* att  = (const float*)d_in[1];  // [T, N_OUT]
    const float* fb   = (const float*)d_in[2];  // [T, N_FB]
    const float* wff  = (const float*)d_in[3];  // [N_IN, N_OUT]
    const float* bias = (const float*)d_in[4];  // [N_OUT]
    const float* wfb  = (const float*)d_in[5];  // [N_FB, N_OUT]
    float* out = (float*)d_out;                 // [T, N_OUT]

    float* FF = (float*)d_ws;                   // [T, N_OUT] f32, 16 MB
    float* FB = FF + (size_t)T * N_OUT;         // [T, N_OUT] f32, 16 MB

    dim3 block(256);
    dim3 grid(N_OUT / TILE_N, T / TILE_M);      // 32 x 8
    gemm_bf16x3<<<grid, block, 0, stream>>>(x,  wff, FF, N_IN, N_OUT);
    gemm_bf16x3<<<grid, block, 0, stream>>>(fb, wfb, FB, N_FB, N_OUT);

    izh_scan<<<N_OUT / 256, block, 0, stream>>>(att, bias, FF, FB, out, T, N_OUT);
}
